// AlignNet_12197707121006
// MI455X (gfx1250) — matrix operand order repaired and checked by reading, not compiled, not run on hardware
//
#include <hip/hip_runtime.h>
#include <hip/hip_bf16.h>

typedef __attribute__((ext_vector_type(16))) _Float16 v16h;
typedef __attribute__((ext_vector_type(8)))  _Float16 v8h;
typedef __attribute__((ext_vector_type(8)))  float    v8f;

#define NFC   64
#define DGN   8
#define KKT   9
#define BATCH 4
#define HH    192
#define WW    320
#define HWSZ  (HH * WW)
#define NP    32    // pixels per workgroup (along W)
#define KCH   576   // K staging chunk (== 64*9)

__device__ __forceinline__ float lrelu_f(float v) { return v >= 0.f ? v : 0.1f * v; }

// Build a 16-wide f16 A/B fragment from two 8-element runs (ISA 16-bit A layout:
// lane half selects K runs {kh8..kh8+7} and {16+kh8..16+kh8+7}).
__device__ __forceinline__ v16h load_a_frag(const _Float16* row, int kk, int kh8) {
    v8h lo = *(const v8h*)(row + kk + kh8);
    v8h hi = *(const v8h*)(row + kk + 16 + kh8);
    return __builtin_shufflevector(lo, hi, 0,1,2,3,4,5,6,7,8,9,10,11,12,13,14,15);
}

// ---------------------------------------------------------------------------
// Generic 3x3 conv (stride 1, pad 1) as implicit-im2col WMMA GEMM.
// x1 supplies channels [64,128) for concat inputs (CIN==128).
// ---------------------------------------------------------------------------
template<int CIN, int COUT, int COUT_PAD, bool RELU>
__global__ __launch_bounds__(256) void conv3x3_wmma(
    const float* __restrict__ x0, const float* __restrict__ x1,
    const _Float16* __restrict__ wpk,   // packed f16 weights [COUT_PAD][CIN*9]
    const float* __restrict__ bias,
    float* __restrict__ out)
{
    constexpr int K     = CIN * 9;
    constexpr int MT    = COUT_PAD / 16;
    constexpr int TILES = MT * 2;               // NP/16 == 2 n-tiles
    constexpr int MAXT  = (TILES + 7) / 8;

    __shared__ __align__(64) _Float16 Bt[NP][KCH];   // im2col tile (f16), 36 KB

    const int wblk = blockIdx.x * NP;
    const int row  = blockIdx.y;                // b*H + h
    const int b    = row / HH, h = row % HH;
    const int t    = threadIdx.x;
    const int wave = t >> 5, lane = t & 31;
    const int kh8  = (lane >> 4) * 8;

    v8f acc[MAXT];
    for (int i = 0; i < MAXT; i++)
        for (int j = 0; j < 8; j++) acc[i][j] = 0.f;

    for (int kc = 0; kc < K; kc += KCH) {
        __syncthreads();
        // ---- stage: im2col chunk [kc, kc+KCH) -> LDS (f32 -> f16) ----
        for (int e = t; e < NP * KCH; e += 256) {
            int kl = e / NP, p = e % NP;
            int k  = kc + kl;
            int ci = k / 9, tap = k % 9;
            int yy = h + tap / 3 - 1;
            int xx = wblk + p + tap % 3 - 1;
            float v = 0.f;
            if ((unsigned)yy < (unsigned)HH && (unsigned)xx < (unsigned)WW) {
                const float* src;
                if (CIN == 128 && ci >= 64)
                    src = x1 + ((size_t)(b * 64 + (ci - 64)) * HH + yy) * WW + xx;
                else
                    src = x0 + ((size_t)(b * 64 + ci) * HH + yy) * WW + xx;
                v = *src;
            }
            Bt[p][kl] = (_Float16)v;
        }
        __syncthreads();

        // ---- GEMM over this chunk ----
        for (int ti = 0; ti < MAXT; ti++) {
            int tile = wave + ti * 8;
            if (tile >= TILES) break;           // wave-uniform
            int mt = tile % MT, nt = tile / MT;
            const _Float16* arow = wpk + (size_t)(mt * 16 + (lane & 15)) * K + kc;
            const _Float16* brow = &Bt[nt * 16 + (lane & 15)][0];
            v8f c = acc[ti];
            for (int kk = 0; kk < KCH; kk += 32) {
                v16h af = load_a_frag(arow, kk, kh8);
                v16h bf = load_a_frag(brow, kk, kh8);
                c = __builtin_amdgcn_wmma_f32_16x16x32_f16(
                        false, af, false, bf, (short)0, c, false, false);
            }
            acc[ti] = c;
        }
    }

    // ---- epilogue: bias (+LeakyReLU) and store fp32 ----
    for (int ti = 0; ti < MAXT; ti++) {
        int tile = wave + ti * 8;
        if (tile >= TILES) break;
        int mt = tile % MT, nt = tile / MT;
        int p  = wblk + nt * 16 + (lane & 15);
        for (int j = 0; j < 8; j++) {
            int o = mt * 16 + j + (lane >> 4) * 8;
            if (o < COUT) {
                float v = acc[ti][j] + bias[o];
                if (RELU) v = lrelu_f(v);
                out[((size_t)(b * COUT + o) * HH + h) * WW + p] = v;
            }
        }
    }
}

// ---------------------------------------------------------------------------
// Modulated deformable conv (DCNv2) as sampled-im2col WMMA GEMM.
// om: [B,224,H,W] fp32 (valid 0..215): [0:144)=offsets (y,x interleaved per
// (g,tap)), [144:216)=mask logits. Output 64ch, LeakyReLU fused.
// ---------------------------------------------------------------------------
__global__ __launch_bounds__(256) void dcn_wmma_kernel(
    const float* __restrict__ x, const float* __restrict__ om,
    const _Float16* __restrict__ wpk,   // packed f16 w_dcn [64][576]
    const float* __restrict__ bias,
    float* __restrict__ out)
{
    constexpr int K = 576;
    __shared__ __align__(64) _Float16 Bt[NP][K];   // 36 KB
    __shared__ float    Ssy[DGN * KKT * NP];       // 9 KB
    __shared__ float    Ssx[DGN * KKT * NP];       // 9 KB
    __shared__ _Float16 Sm [DGN * KKT * NP];       // 4.5 KB

    const int wblk = blockIdx.x * NP;
    const int row  = blockIdx.y;
    const int b    = row / HH, h = row % HH;
    const int t    = threadIdx.x;
    const int wave = t >> 5, lane = t & 31;
    const int kh8  = (lane >> 4) * 8;

    // ---- stage 0: sampling params per (g, tap, pixel) ----
    for (int e = t; e < DGN * KKT * NP; e += 256) {
        int gk = e / NP, p = e % NP;
        int tap = gk % KKT;
        size_t base = (size_t)b * 224 * HWSZ + (size_t)h * WW + (wblk + p);
        float oy = om[base + (size_t)(2 * gk) * HWSZ];
        float ox = om[base + (size_t)(2 * gk + 1) * HWSZ];
        float mr = om[base + (size_t)(144 + gk) * HWSZ];
        Ssy[e] = (float)(h + tap / 3 - 1) + oy;
        Ssx[e] = (float)(wblk + p + tap % 3 - 1) + ox;
        Sm[e]  = (_Float16)(1.0f / (1.0f + __expf(-mr)));
    }
    __syncthreads();

    // ---- stage 1: bilinear sample * mask -> f16 tile ----
    for (int e = t; e < NP * K; e += 256) {
        int k = e / NP, p = e % NP;
        int ci = k / 9, tap = k % 9;
        int si = ((ci >> 3) * KKT + tap) * NP + p;
        float sy = Ssy[si], sx = Ssx[si];
        float msk = (float)Sm[si];
        float y0f = floorf(sy), x0f = floorf(sx);
        int   y0  = (int)y0f,   x0i = (int)x0f;
        float ly = sy - y0f, lx = sx - x0f;
        const float* img = x + (size_t)(b * 64 + ci) * HWSZ;
        float v00 = 0.f, v01 = 0.f, v10 = 0.f, v11 = 0.f;
        bool yin0 = (unsigned)y0        < (unsigned)HH;
        bool yin1 = (unsigned)(y0 + 1)  < (unsigned)HH;
        bool xin0 = (unsigned)x0i       < (unsigned)WW;
        bool xin1 = (unsigned)(x0i + 1) < (unsigned)WW;
        if (yin0 && xin0) v00 = img[y0 * WW + x0i];
        if (yin0 && xin1) v01 = img[y0 * WW + x0i + 1];
        if (yin1 && xin0) v10 = img[(y0 + 1) * WW + x0i];
        if (yin1 && xin1) v11 = img[(y0 + 1) * WW + x0i + 1];
        float v = ((1.f - ly) * ((1.f - lx) * v00 + lx * v01) +
                   ly        * ((1.f - lx) * v10 + lx * v11)) * msk;
        Bt[p][k] = (_Float16)v;
    }
    __syncthreads();

    // ---- GEMM: 64 out-ch, 8 waves = 4 m-tiles x 2 n-tiles ----
    int mt = wave & 3, nt = wave >> 2;
    const _Float16* arow = wpk + (size_t)(mt * 16 + (lane & 15)) * K;
    const _Float16* brow = &Bt[nt * 16 + (lane & 15)][0];
    v8f c;
    for (int j = 0; j < 8; j++) c[j] = 0.f;
    for (int kk = 0; kk < K; kk += 32) {
        v16h af = load_a_frag(arow, kk, kh8);
        v16h bf = load_a_frag(brow, kk, kh8);
        c = __builtin_amdgcn_wmma_f32_16x16x32_f16(
                false, af, false, bf, (short)0, c, false, false);
    }
    int p = wblk + nt * 16 + (lane & 15);
    for (int j = 0; j < 8; j++) {
        int o = mt * 16 + j + (lane >> 4) * 8;
        float v = lrelu_f(c[j] + bias[o]);
        out[((size_t)(b * 64 + o) * HH + h) * WW + p] = v;
    }
}

// fp32 OIHW -> packed f16 [COUT_PAD][CIN*9]; padded rows zeroed.
__global__ void pack_weights_kernel(const float* __restrict__ w,
                                    _Float16* __restrict__ dst,
                                    int valid, int total)
{
    int i = blockIdx.x * 256 + threadIdx.x;
    if (i < total) dst[i] = (i < valid) ? (_Float16)w[i] : (_Float16)0.f;
}

// ---------------------------------------------------------------------------
extern "C" void kernel_launch(void* const* d_in, const int* in_sizes, int n_in,
                              void* d_out, int out_size, void* d_ws, size_t ws_size,
                              hipStream_t stream)
{
    (void)in_sizes; (void)n_in; (void)out_size; (void)ws_size;

    const float* fea[5];
    for (int i = 0; i < 5; i++) fea[i] = (const float*)d_in[i];
    const float* w_of1 = (const float*)d_in[5];  const float* b_of1 = (const float*)d_in[6];
    const float* w_of2 = (const float*)d_in[7];  const float* b_of2 = (const float*)d_in[8];
    const float* w_or1 = (const float*)d_in[9];  const float* b_or1 = (const float*)d_in[10];
    const float* w_or2 = (const float*)d_in[11]; const float* b_or2 = (const float*)d_in[12];
    const float* w_om  = (const float*)d_in[13]; const float* b_om  = (const float*)d_in[14];
    const float* w_dcn = (const float*)d_in[15]; const float* b_dcn = (const float*)d_in[16];
    const float* w_f1  = (const float*)d_in[17]; const float* b_f1  = (const float*)d_in[18];
    const float* w_f2  = (const float*)d_in[19]; const float* b_f2  = (const float*)d_in[20];

    // ---- workspace carve-up (256B aligned) ----
    size_t off = 0;
    auto alloc = [&](size_t bytes) -> void* {
        void* p = (void*)((char*)d_ws + off);
        off += (bytes + 255) & ~(size_t)255;
        return p;
    };
    const size_t ACT = (size_t)BATCH * 64 * HWSZ * sizeof(float);
    float* t0    = (float*)alloc(ACT);
    float* ofb   = (float*)alloc(ACT);
    float* dfea  = (float*)alloc(ACT);
    float* f1b   = (float*)alloc(ACT);
    float* leftb = (float*)alloc(ACT);
    float* rightb= (float*)alloc(ACT);
    float* omb   = (float*)alloc((size_t)BATCH * 224 * HWSZ * sizeof(float));
    _Float16* pw_of1 = (_Float16*)alloc((size_t)64 * 1152 * 2);
    _Float16* pw_of2 = (_Float16*)alloc((size_t)64 * 576 * 2);
    _Float16* pw_or1 = (_Float16*)alloc((size_t)64 * 1152 * 2);
    _Float16* pw_or2 = (_Float16*)alloc((size_t)64 * 576 * 2);
    _Float16* pw_om  = (_Float16*)alloc((size_t)224 * 576 * 2);
    _Float16* pw_dcn = (_Float16*)alloc((size_t)64 * 576 * 2);
    _Float16* pw_f1  = (_Float16*)alloc((size_t)64 * 1152 * 2);
    _Float16* pw_f2  = (_Float16*)alloc((size_t)64 * 576 * 2);

    auto pack = [&](const float* src, _Float16* dst, int valid, int total) {
        pack_weights_kernel<<<(total + 255) / 256, 256, 0, stream>>>(src, dst, valid, total);
    };
    pack(w_of1, pw_of1, 64 * 1152, 64 * 1152);
    pack(w_of2, pw_of2, 64 * 576,  64 * 576);
    pack(w_or1, pw_or1, 64 * 1152, 64 * 1152);
    pack(w_or2, pw_or2, 64 * 576,  64 * 576);
    pack(w_om,  pw_om,  216 * 576, 224 * 576);
    pack(w_dcn, pw_dcn, 64 * 576,  64 * 576);
    pack(w_f1,  pw_f1,  64 * 1152, 64 * 1152);
    pack(w_f2,  pw_f2,  64 * 576,  64 * 576);

    dim3 grid(WW / NP, BATCH * HH);

    auto conv128 = [&](const float* a, const float* b2, const _Float16* wp,
                       const float* bi, float* o) {
        conv3x3_wmma<128, 64, 64, true><<<grid, 256, 0, stream>>>(a, b2, wp, bi, o);
    };
    auto conv64 = [&](const float* a, const _Float16* wp, const float* bi, float* o) {
        conv3x3_wmma<64, 64, 64, true><<<grid, 256, 0, stream>>>(a, nullptr, wp, bi, o);
    };
    auto convom = [&](const float* a, float* o) {
        conv3x3_wmma<64, 216, 224, false><<<grid, 256, 0, stream>>>(a, nullptr, pw_om, b_om, o);
    };
    auto dcnk = [&](const float* xin, float* o) {
        dcn_wmma_kernel<<<grid, 256, 0, stream>>>(xin, omb, pw_dcn, b_dcn, o);
    };

    auto align_block = [&](const float* feaA, const float* refA, float* outA) {
        conv128(feaA, refA, pw_of1, b_of1, t0);     // lrelu(conv(cat(fea,ref)))
        conv64(t0, pw_of2, b_of2, ofb);             // lrelu(conv(of))
        convom(ofb, omb);                           // offsets/mask logits
        dcnk(feaA, dfea);                           // dfea = lrelu(dcn(fea))
        conv128(refA, feaA, pw_or1, b_or1, t0);
        conv64(t0, pw_or2, b_or2, ofb);
        convom(ofb, omb);
        dcnk(refA, t0);                             // dref = lrelu(dcn(ref)) -> t0
        conv128(dfea, t0, pw_f1, b_f1, ofb);        // fusion conv1
        conv64(ofb, pw_f2, b_f2, outA);             // fusion conv2 -> out
    };

    align_block(fea[0], fea[1], f1b);
    align_block(f1b, fea[2], leftb);
    align_block(fea[4], fea[3], rightb);
    align_block(leftb, rightb, (float*)d_out);
}
